// ContextEncoder_50929722196118
// MI455X (gfx1250) — compile-verified, matrix operand order checked by reading, not run on hardware
//
#include <hip/hip_runtime.h>
#include <math.h>

// ---------------------------------------------------------------------------
// Problem constants (from the reference)
// ---------------------------------------------------------------------------
constexpr int Bb = 4, Tt = 2048, Dd = 768, Hh = 8, HDd = 96, Ff = 2048, Ll = 2;
constexpr int MTOK = Bb * Tt;          // 8192 token rows
constexpr float EPSF = 1e-6f;

typedef __bf16 bf16_t;
typedef __attribute__((ext_vector_type(16))) __bf16 v16bf;
typedef __attribute__((ext_vector_type(8)))  float  v8f;
typedef unsigned int u32x4 __attribute__((ext_vector_type(4)));
typedef int          i32x4 __attribute__((ext_vector_type(4)));
typedef int          i32x8 __attribute__((ext_vector_type(8)));

union Frag { v16bf v; unsigned int u[8]; };

// ---------------------------------------------------------------------------
// TDM: issue a 2D tensor-tile load (global -> LDS) via the Tensor Data Mover.
// Descriptor packing per CDNA5 ISA ch.8 (D# group0/group1; groups 2-3 zero for
// 2D). Hardware zero-fills columns beyond tensor_dim0 (edge tiles) and inserts
// LDS row padding (pad_interval/pad_amount) so fragment reads stay
// bank-conflict free. Tracked by TENSORcnt.
// ---------------------------------------------------------------------------
__device__ inline void tdm_load_2d(unsigned lds_addr, const bf16_t* gaddr,
                                   int width, int rows, int stride, int tile_w,
                                   int pad_iv) {
  const unsigned long long ga = (unsigned long long)gaddr;
  u32x4 g0;
  g0[0] = 0x1u;                                   // count=1, user descriptor
  g0[1] = lds_addr;                               // LDS byte address
  g0[2] = (unsigned)(ga & 0xffffffffu);           // global addr [31:0]
  g0[3] = (unsigned)((ga >> 32) & 0x01ffffffu) | (2u << 30);  // addr[56:32], type=2

  i32x8 g1;
  // data_size=2B (code 1), pad_enable, pad_interval, pad_amount=1 dword (code 0)
  g1[0] = (int)((1u << 16) | (1u << 20) | ((unsigned)pad_iv << 22));
  g1[1] = (int)(((unsigned)width & 0xffffu) << 16);                // tensor_dim0[15:0]
  g1[2] = (int)((((unsigned)width >> 16) & 0xffffu) |
                (((unsigned)rows & 0xffffu) << 16));               // dim0[31:16] | dim1[15:0]
  g1[3] = (int)((((unsigned)rows >> 16) & 0xffffu) |
                (((unsigned)tile_w & 0xffffu) << 16));             // dim1[31:16] | tile_dim0
  g1[4] = (int)((unsigned)rows & 0xffffu);                         // tile_dim1 (tile_dim2=0)
  g1[5] = (int)(unsigned)stride;                                   // dim0_stride[31:0]
  g1[6] = 0;                                                       // stride hi bits
  g1[7] = 0;
  i32x4 gz = {0, 0, 0, 0};
#if __clang_major__ >= 23
  i32x8 gz8 = {0, 0, 0, 0, 0, 0, 0, 0};
  __builtin_amdgcn_tensor_load_to_lds(g0, g1, gz, gz, gz8, 0);
#else
  __builtin_amdgcn_tensor_load_to_lds(g0, g1, gz, gz, 0);
#endif
}

// ---------------------------------------------------------------------------
// f32 -> bf16 cast
// ---------------------------------------------------------------------------
__global__ void cast_bf16_kernel(const float* __restrict__ x, bf16_t* __restrict__ y, size_t n) {
  size_t i = (size_t)blockIdx.x * blockDim.x + threadIdx.x;
  if (i < n) y[i] = (bf16_t)x[i];
}

// ---------------------------------------------------------------------------
// LayerNorm over D=768 (one 256-thread block per row), optional mask,
// writes f32 and bf16 (bf16 feeds the WMMA GEMMs).
// ---------------------------------------------------------------------------
__global__ void layernorm_kernel(const float* __restrict__ x, const float* __restrict__ w,
                                 const float* __restrict__ b, const float* __restrict__ mask,
                                 float* __restrict__ yf, bf16_t* __restrict__ yb) {
  __shared__ float s1[256], s2[256];
  const int row = blockIdx.x, tid = threadIdx.x;
  const float* xr = x + (size_t)row * Dd;
  float v0 = xr[tid], v1 = xr[tid + 256], v2 = xr[tid + 512];
  s1[tid] = v0 + v1 + v2;
  s2[tid] = v0 * v0 + v1 * v1 + v2 * v2;
  __syncthreads();
  for (int st = 128; st > 0; st >>= 1) {
    if (tid < st) { s1[tid] += s1[tid + st]; s2[tid] += s2[tid + st]; }
    __syncthreads();
  }
  const float mean = s1[0] * (1.0f / Dd);
  const float var  = s2[0] * (1.0f / Dd) - mean * mean;
  const float inv  = rsqrtf(var + 1e-5f);
  const float mk   = mask ? mask[row] : 1.0f;
#pragma unroll
  for (int j = 0; j < 3; j++) {
    int c = tid + j * 256;
    float y = ((xr[c] - mean) * inv * w[c] + b[c]) * mk;
    yf[(size_t)row * Dd + c] = y;
    if (yb) yb[(size_t)row * Dd + c] = (bf16_t)y;
  }
}

// ---------------------------------------------------------------------------
// WMMA bf16 GEMM:  C[M,N] = act(A[M,K] * B[K,N] + bias) + resid
// Block tile 128x128, BK=32, 256 threads = 8 waves (2x4), wave tile 64x32:
// 4 A-frags x 2 B-frags = 8 WMMAs per K-step.
// Double-buffered LDS staged by the Tensor Data Mover: wave 0 issues the
// descriptors for tile i+1, then s_wait_tensorcnt 2 (per-wave in-order TDM
// completion => tile i has landed) so the DMA for i+1 overlaps the WMMAs of
// tile i. Hardware pads LDS rows to an odd dword stride and zero-fills N-edge
// columns, so there is no staging code and no divergent edge path.
//   A in LDS: 128 x 32 bf16, row stride 34 elems (pad 1 dword / 16 dwords)
//   B in LDS:  32 x 128 bf16, row stride 130 elems (pad 1 dword / 64 dwords)
// Fragment layout per CDNA5 ISA 7.12.2 (16-bit A 16x32, mirrored B).
// ---------------------------------------------------------------------------
__global__ __launch_bounds__(256)
void gemm_bf16_wmma(const bf16_t* __restrict__ A, const bf16_t* __restrict__ Bm,
                    const float* __restrict__ bias, const float* __restrict__ resid,
                    float* __restrict__ C, bf16_t* __restrict__ Cb,
                    int M, int N, int K, int act) {
  __shared__ bf16_t As[2][128 * 34];   // 2 x 8704 B
  __shared__ bf16_t Bs[2][32 * 130];   // 2 x 8320 B

  const int tid  = threadIdx.x;
  const int m0   = blockIdx.y * 128;
  const int n0   = blockIdx.x * 128;
  const int lane = tid & 31, wave = tid >> 5;
  const int wr = wave >> 2, wc = wave & 3;         // 2x4 wave grid, wave tile 64x32
  const int half = lane >> 4, mrow = lane & 15;

  v8f acc[4][2] = {};

  const int nt = K >> 5;               // K-tiles of 32
  if (wave == 0) {
    tdm_load_2d((unsigned)(size_t)&As[0][0], A + (size_t)m0 * K, 32, 128, K, 32, 3);
    tdm_load_2d((unsigned)(size_t)&Bs[0][0], Bm + n0, N - n0, 32, N, 128, 5);
  }

  for (int i = 0; i < nt; i++) {
    const int cur = i & 1;
    if (wave == 0) {
      if (i + 1 < nt) {
        const int k0 = (i + 1) << 5;
        tdm_load_2d((unsigned)(size_t)&As[cur ^ 1][0], A + (size_t)m0 * K + k0,
                    32, 128, K, 32, 3);
        tdm_load_2d((unsigned)(size_t)&Bs[cur ^ 1][0], Bm + (size_t)k0 * N + n0,
                    N - n0, 32, N, 128, 5);
        __builtin_amdgcn_s_wait_tensorcnt(2);   // tile i complete (in-order)
      } else {
        __builtin_amdgcn_s_wait_tensorcnt(0);
      }
    }
    __syncthreads();

    const bf16_t* Asb = As[cur];
    const unsigned short* Bsu = (const unsigned short*)Bs[cur];

    Frag a[4], b[2];
#pragma unroll
    for (int v = 0; v < 8; v++) {
      const int kf = ((v & 3) << 1) + (half << 3) + ((v >> 2) << 4);
#pragma unroll
      for (int f = 0; f < 4; f++)
        a[f].u[v] = *(const unsigned int*)&Asb[(wr * 64 + f * 16 + mrow) * 34 + kf];
#pragma unroll
      for (int j = 0; j < 2; j++) {
        const int nn = wc * 32 + j * 16 + mrow;
        b[j].u[v] = (unsigned)Bsu[kf * 130 + nn] |
                    ((unsigned)Bsu[(kf + 1) * 130 + nn] << 16);
      }
    }
#pragma unroll
    for (int f = 0; f < 4; f++) {
#pragma unroll
      for (int j = 0; j < 2; j++) {
        acc[f][j] = __builtin_amdgcn_wmma_f32_16x16x32_bf16(
            false, a[f].v, false, b[j].v, (short)0, acc[f][j], false, false);
      }
    }
    __syncthreads();   // buf[cur] free for the i+2 prefetch
  }

  // ---- epilogue: bias / SiLU / residual, f32 + bf16 stores ----
#pragma unroll
  for (int f = 0; f < 4; f++) {
#pragma unroll
    for (int j = 0; j < 2; j++) {
      const int mb = m0 + wr * 64 + f * 16 + half * 8;
      const int nn = n0 + wc * 32 + j * 16 + mrow;
      if (nn < N) {
#pragma unroll
        for (int r = 0; r < 8; r++) {
          float vv = acc[f][j][r];
          if (bias)  vv += bias[nn];
          if (act == 1) vv = vv * (1.0f / (1.0f + __expf(-vv)));   // SiLU
          const size_t idx = (size_t)(mb + r) * N + nn;
          if (resid) vv += resid[idx];
          if (C)  C[idx]  = vv;
          if (Cb) Cb[idx] = (bf16_t)vv;
        }
      }
    }
  }
}

// ---------------------------------------------------------------------------
// L2-norm over HD=96 per (token, head), in place. One wave per head-row.
// ---------------------------------------------------------------------------
__global__ void l2norm_kernel(float* __restrict__ q, float scale, int nheads) {
  const int gw = (int)((blockIdx.x * (size_t)blockDim.x + threadIdx.x) >> 5);
  const int lane = threadIdx.x & 31;
  if (gw >= nheads) return;
  float* p = q + (size_t)gw * 32 * 3;     // HD = 96 contiguous
  float a = p[lane], b = p[lane + 32], c = p[lane + 64];
  float ss = a * a + b * b + c * c;
#pragma unroll
  for (int m = 16; m > 0; m >>= 1) ss += __shfl_xor(ss, m, 32);
  const float inv = rsqrtf(ss + EPSF) * scale;
  p[lane] = a * inv; p[lane + 32] = b * inv; p[lane + 64] = c * inv;
}

// ---------------------------------------------------------------------------
// g = -exp(A_log[j]) * softplus(f[i,j])   (f already has dt_bias from GEMM)
// ---------------------------------------------------------------------------
__global__ void g_kernel(const float* __restrict__ f, const float* __restrict__ A_log,
                         float* __restrict__ g, size_t n) {
  size_t i = (size_t)blockIdx.x * blockDim.x + threadIdx.x;
  if (i >= n) return;
  const int j = (int)(i % (Hh * HDd));
  const float x = f[i];
  const float sp = (x > 20.0f) ? x : log1pf(__expf(x));
  g[i] = -__expf(A_log[j]) * sp;
}

// ---------------------------------------------------------------------------
// beta = sigmoid(x @ Wb), N=8 too small for WMMA: one thread per (token, h)
// ---------------------------------------------------------------------------
__global__ void beta_kernel(const float* __restrict__ x, const float* __restrict__ Wb,
                            float* __restrict__ beta) {
  const int idx = blockIdx.x * blockDim.x + threadIdx.x;
  if (idx >= MTOK * Hh) return;
  const int tkn = idx / Hh, h = idx % Hh;
  const float* xr = x + (size_t)tkn * Dd;
  float s = 0.0f;
#pragma unroll 4
  for (int i = 0; i < Dd; i++) s += xr[i] * Wb[i * Hh + h];
  beta[idx] = 1.0f / (1.0f + __expf(-s));
}

// ---------------------------------------------------------------------------
// Delta-rule recurrence. One block per (b,h); 96x96 f32 state in LDS
// (stride 97 to avoid bank conflicts); 96 threads own one value-column each.
//   S = S * exp(g)[k];  kv = k^T S;  delta = (v - kv)*beta;
//   S += k delta^T;     o = q^T S
// ---------------------------------------------------------------------------
__global__ __launch_bounds__(128)
void scan_kernel(const float* __restrict__ q, const float* __restrict__ k,
                 const float* __restrict__ v, const float* __restrict__ g,
                 const float* __restrict__ beta,
                 float* __restrict__ o, float* __restrict__ Sout) {
  __shared__ float S[HDd * 97];
  __shared__ float kv_[HDd], qv_[HDd], vv_[HDd], eg_[HDd];
  __shared__ float bsh;
  const int tid = threadIdx.x;
  const int b = blockIdx.x / Hh, h = blockIdx.x % Hh;

  for (int i = tid; i < HDd * 97; i += blockDim.x) S[i] = 0.0f;
  __syncthreads();

  for (int t = 0; t < Tt; t++) {
    const size_t base = ((size_t)(b * Tt + t)) * (Hh * HDd) + (size_t)h * HDd;
    if (tid < HDd) {
      kv_[tid] = k[base + tid];
      qv_[tid] = q[base + tid];
      vv_[tid] = v[base + tid];
      eg_[tid] = __expf(g[base + tid]);
    }
    if (tid == 0) bsh = beta[(size_t)(b * Tt + t) * Hh + h];
    __syncthreads();
    if (tid < HDd) {
      const int c = tid;
      float kv = 0.0f;
#pragma unroll 4
      for (int kk = 0; kk < HDd; kk++) {
        float s = S[kk * 97 + c] * eg_[kk];
        S[kk * 97 + c] = s;
        kv += kv_[kk] * s;
      }
      const float delta = (vv_[c] - kv) * bsh;
      float out = 0.0f;
#pragma unroll 4
      for (int kk = 0; kk < HDd; kk++) {
        float s = S[kk * 97 + c] + kv_[kk] * delta;
        S[kk * 97 + c] = s;
        out += qv_[kk] * s;
      }
      o[base + c] = out;
    }
    __syncthreads();
  }
  for (int i = tid; i < HDd * HDd; i += blockDim.x) {
    const int kk = i / HDd, c = i % HDd;
    Sout[(size_t)blockIdx.x * HDd * HDd + i] = S[kk * 97 + c];
  }
}

// ---------------------------------------------------------------------------
// Gated RMSNorm per head: o = o * rsqrt(mean(o^2)+eps) * onorm_w * silu(gate)
// One wave per (token, head); writes bf16 for the Wo GEMM.
// ---------------------------------------------------------------------------
__global__ void onorm_gate_kernel(const float* __restrict__ o, const float* __restrict__ gate,
                                  const float* __restrict__ onw, bf16_t* __restrict__ ob,
                                  int nheads) {
  const int gw = (int)((blockIdx.x * (size_t)blockDim.x + threadIdx.x) >> 5);
  const int lane = threadIdx.x & 31;
  if (gw >= nheads) return;
  const size_t base = (size_t)gw * HDd;
  float a = o[base + lane], b = o[base + lane + 32], c = o[base + lane + 64];
  float ss = a * a + b * b + c * c;
#pragma unroll
  for (int m = 16; m > 0; m >>= 1) ss += __shfl_xor(ss, m, 32);
  const float inv = rsqrtf(ss * (1.0f / HDd) + EPSF);
#pragma unroll
  for (int j = 0; j < 3; j++) {
    const int hd = lane + 32 * j;
    const float gt = gate[base + hd];
    float val = o[base + hd] * inv * onw[hd];
    val *= gt * (1.0f / (1.0f + __expf(-gt)));   // silu(gate)
    ob[base + hd] = (bf16_t)val;
  }
}

// ---------------------------------------------------------------------------
// m = g1 * u  (g1 already SiLU'd in its GEMM), emitted as bf16 for down-GEMM
// ---------------------------------------------------------------------------
__global__ void mul_bf16_kernel(const float* __restrict__ a, const float* __restrict__ b,
                                bf16_t* __restrict__ y, size_t n) {
  size_t i = (size_t)blockIdx.x * blockDim.x + threadIdx.x;
  if (i < n) y[i] = (bf16_t)(a[i] * b[i]);
}

// ---------------------------------------------------------------------------
// Host-side orchestration
// ---------------------------------------------------------------------------
static inline void cast_w(const float* src, bf16_t* dst, size_t n, hipStream_t s) {
  cast_bf16_kernel<<<(unsigned)((n + 255) / 256), 256, 0, s>>>(src, dst, n);
}
static inline void gemm(const bf16_t* A, const bf16_t* Bm, const float* bias,
                        const float* resid, float* C, bf16_t* Cb,
                        int M, int N, int K, int act, hipStream_t s) {
  dim3 g((N + 127) / 128, M / 128);
  gemm_bf16_wmma<<<g, 256, 0, s>>>(A, Bm, bias, resid, C, Cb, M, N, K, act);
}

extern "C" void kernel_launch(void* const* d_in, const int* in_sizes, int n_in,
                              void* d_out, int out_size, void* d_ws, size_t ws_size,
                              hipStream_t stream) {
  (void)in_sizes; (void)n_in; (void)out_size; (void)ws_size;
  const float* hidden  = (const float*)d_in[0];
  const float* amask   = (const float*)d_in[1];
  const float* norm_aw = (const float*)d_in[2];
  const float* norm_ab = (const float*)d_in[3];
  const float* proj_aw = (const float*)d_in[4];
  const float* proj_ab = (const float*)d_in[5];
  const float* ln_w    = (const float*)d_in[6];
  const float* ln_b    = (const float*)d_in[7];
  const float* Wq      = (const float*)d_in[8];
  const float* Wk      = (const float*)d_in[9];
  const float* Wv      = (const float*)d_in[10];
  const float* Wfa     = (const float*)d_in[11];
  const float* Wfb     = (const float*)d_in[12];
  const float* dt_bias = (const float*)d_in[13];
  const float* A_log   = (const float*)d_in[14];
  const float* Wb      = (const float*)d_in[15];
  const float* Wga     = (const float*)d_in[16];
  const float* Wgb     = (const float*)d_in[17];
  const float* onorm_w = (const float*)d_in[18];
  const float* Wo      = (const float*)d_in[19];
  const float* Wgate   = (const float*)d_in[20];
  const float* Wup     = (const float*)d_in[21];
  const float* Wdown   = (const float*)d_in[22];

  float* out_h = (float*)d_out;                       // [B,T,D]
  float* out_S = out_h + (size_t)MTOK * Dd;           // [L,B,H,HD,HD]

  // bump allocator over d_ws
  size_t off = 0;
  auto alloc = [&](size_t bytes) -> void* {
    bytes = (bytes + 255) & ~(size_t)255;
    void* p = (char*)d_ws + off;
    off += bytes;
    return p;
  };
  const size_t MD = (size_t)MTOK * Dd;

  // bf16 weight mirrors
  bf16_t* pa_bf    = (bf16_t*)alloc((size_t)Dd * Dd * 2);
  bf16_t* wq_bf    = (bf16_t*)alloc((size_t)Ll * Dd * Dd * 2);
  bf16_t* wk_bf    = (bf16_t*)alloc((size_t)Ll * Dd * Dd * 2);
  bf16_t* wv_bf    = (bf16_t*)alloc((size_t)Ll * Dd * Dd * 2);
  bf16_t* wfa_bf   = (bf16_t*)alloc((size_t)Ll * Dd * HDd * 2);
  bf16_t* wfb_bf   = (bf16_t*)alloc((size_t)Ll * HDd * Dd * 2);
  bf16_t* wga_bf   = (bf16_t*)alloc((size_t)Ll * Dd * HDd * 2);
  bf16_t* wgb_bf   = (bf16_t*)alloc((size_t)Ll * HDd * Dd * 2);
  bf16_t* wo_bf    = (bf16_t*)alloc((size_t)Ll * Dd * Dd * 2);
  bf16_t* wgate_bf = (bf16_t*)alloc((size_t)Ll * Dd * Ff * 2);
  bf16_t* wup_bf   = (bf16_t*)alloc((size_t)Ll * Dd * Ff * 2);
  bf16_t* wdown_bf = (bf16_t*)alloc((size_t)Ll * Ff * Dd * 2);

  // activations
  float*  xf   = (float*)alloc(MD * 4);
  bf16_t* xb   = (bf16_t*)alloc(MD * 2);
  float*  h0   = (float*)alloc(MD * 4);
  float*  h1   = (float*)alloc(MD * 4);
  float*  qf   = (float*)alloc(MD * 4);
  float*  kf   = (float*)alloc(MD * 4);
  float*  vf   = (float*)alloc(MD * 4);
  bf16_t* fab  = (bf16_t*)alloc((size_t)MTOK * HDd * 2);
  float*  ff   = (float*)alloc(MD * 4);
  float*  betaf= (float*)alloc((size_t)MTOK * Hh * 4);
  float*  of   = (float*)alloc(MD * 4);
  bf16_t* gab  = (bf16_t*)alloc((size_t)MTOK * HDd * 2);
  float*  ggf  = (float*)alloc(MD * 4);
  bf16_t* ob   = (bf16_t*)alloc(MD * 2);
  bf16_t* ab   = (bf16_t*)alloc(MD * 2);
  float*  g1f  = (float*)alloc((size_t)MTOK * Ff * 4);
  float*  uf   = (float*)alloc((size_t)MTOK * Ff * 4);
  bf16_t* mb   = (bf16_t*)alloc((size_t)MTOK * Ff * 2);

  // ---- weights -> bf16 ----
  cast_w(proj_aw, pa_bf,    (size_t)Dd * Dd, stream);
  cast_w(Wq,      wq_bf,    (size_t)Ll * Dd * Dd, stream);
  cast_w(Wk,      wk_bf,    (size_t)Ll * Dd * Dd, stream);
  cast_w(Wv,      wv_bf,    (size_t)Ll * Dd * Dd, stream);
  cast_w(Wfa,     wfa_bf,   (size_t)Ll * Dd * HDd, stream);
  cast_w(Wfb,     wfb_bf,   (size_t)Ll * HDd * Dd, stream);
  cast_w(Wga,     wga_bf,   (size_t)Ll * Dd * HDd, stream);
  cast_w(Wgb,     wgb_bf,   (size_t)Ll * HDd * Dd, stream);
  cast_w(Wo,      wo_bf,    (size_t)Ll * Dd * Dd, stream);
  cast_w(Wgate,   wgate_bf, (size_t)Ll * Dd * Ff, stream);
  cast_w(Wup,     wup_bf,   (size_t)Ll * Dd * Ff, stream);
  cast_w(Wdown,   wdown_bf, (size_t)Ll * Ff * Dd, stream);

  const int nheads = MTOK * Hh;                       // 65536 (token, head) rows
  const unsigned wavesGrid = (unsigned)((nheads + 7) / 8); // 8 waves / 256-thr block
  const float qscale = 1.0f / sqrtf((float)HDd);

  // ---- stem: h = layernorm(hidden) @ proj_a + bias ----
  layernorm_kernel<<<MTOK, 256, 0, stream>>>(hidden, norm_aw, norm_ab, nullptr, xf, xb);
  gemm(xb, pa_bf, proj_ab, nullptr, h0, nullptr, MTOK, Dd, Dd, 0, stream);

  float* hcur = h0;
  for (int l = 0; l < Ll; l++) {
    float* hnext = (l == Ll - 1) ? out_h : h1;
    const size_t wDD = (size_t)l * Dd * Dd, wDH = (size_t)l * Dd * HDd;
    const size_t wHD2 = (size_t)l * HDd * Dd, wDF = (size_t)l * Dd * Ff;

    // x = mask * layernorm(h)
    layernorm_kernel<<<MTOK, 256, 0, stream>>>(hcur, ln_w + (size_t)l * Dd,
                                               ln_b + (size_t)l * Dd, amask, xf, xb);

    // q/k/v = silu(x @ W*) ; then l2norm(q)*HD^-0.5, l2norm(k)
    gemm(xb, wq_bf + wDD, nullptr, nullptr, qf, nullptr, MTOK, Dd, Dd, 1, stream);
    gemm(xb, wk_bf + wDD, nullptr, nullptr, kf, nullptr, MTOK, Dd, Dd, 1, stream);
    gemm(xb, wv_bf + wDD, nullptr, nullptr, vf, nullptr, MTOK, Dd, Dd, 1, stream);
    l2norm_kernel<<<wavesGrid, 256, 0, stream>>>(qf, qscale, nheads);
    l2norm_kernel<<<wavesGrid, 256, 0, stream>>>(kf, 1.0f, nheads);

    // g = -exp(A_log) * softplus((x@Wfa)@Wfb + dt_bias)
    gemm(xb, wfa_bf + wDH, nullptr, nullptr, nullptr, fab, MTOK, HDd, Dd, 0, stream);
    gemm(fab, wfb_bf + wHD2, dt_bias + (size_t)l * Dd, nullptr, ff, nullptr, MTOK, Dd, HDd, 0, stream);
    g_kernel<<<(unsigned)((MD + 255) / 256), 256, 0, stream>>>(ff, A_log + (size_t)l * Dd, ff, MD);

    // beta = sigmoid(x @ Wb)
    beta_kernel<<<(MTOK * Hh + 255) / 256, 256, 0, stream>>>(xf, Wb + (size_t)l * Dd * Hh, betaf);

    // sequential delta-rule scan; writes o and the layer's final state
    scan_kernel<<<Bb * Hh, 128, 0, stream>>>(qf, kf, vf, ff, betaf, of,
                                             out_S + (size_t)l * Bb * Hh * HDd * HDd);

    // gate = (x@Wga)@Wgb ; o = rmsnorm(o)*onorm_w*silu(gate)
    gemm(xb, wga_bf + wDH, nullptr, nullptr, nullptr, gab, MTOK, HDd, Dd, 0, stream);
    gemm(gab, wgb_bf + wHD2, nullptr, nullptr, ggf, nullptr, MTOK, Dd, HDd, 0, stream);
    onorm_gate_kernel<<<wavesGrid, 256, 0, stream>>>(of, ggf, onorm_w + (size_t)l * HDd, ob, nheads);

    // a = o @ Wo
    gemm(ob, wo_bf + wDD, nullptr, nullptr, nullptr, ab, MTOK, Dd, Dd, 0, stream);

    // SwiGLU MLP with fused residual on the down projection
    gemm(ab, wgate_bf + wDF, nullptr, nullptr, g1f, nullptr, MTOK, Ff, Dd, 1, stream);
    gemm(ab, wup_bf + wDF,   nullptr, nullptr, uf,  nullptr, MTOK, Ff, Dd, 0, stream);
    mul_bf16_kernel<<<(unsigned)(((size_t)MTOK * Ff + 255) / 256), 256, 0, stream>>>(
        g1f, uf, mb, (size_t)MTOK * Ff);
    gemm(mb, wdown_bf + (size_t)l * Ff * Dd, nullptr, hcur, hnext, nullptr, MTOK, Dd, Ff, 0, stream);

    hcur = hnext;
  }
}
